// CrossModalTransformer_49701361550109
// MI455X (gfx1250) — matrix operand order repaired and checked by reading, not compiled, not run on hardware
//
#include <hip/hip_runtime.h>
#include <hip/hip_bf16.h>

// ---------------------------------------------------------------------------
// CrossModalTransformer on gfx1250 (MI455X), wave32 + WMMA f16 path.
// Dead FFN skipped. Flash-style attention with async-to-LDS double buffering.
// ---------------------------------------------------------------------------

typedef _Float16 f16;
typedef __attribute__((ext_vector_type(16))) _Float16 v16h;
typedef __attribute__((ext_vector_type(8)))  _Float16 h8;
typedef __attribute__((ext_vector_type(8)))  float    v8f;

#define BS 4
#define S1 2048
#define S2 2048
#define DMODEL 512
#define HEADS 8
#define HDIM 64
#define NEGV (-1e23f)

__device__ __forceinline__ v8f wmma_f16(v16h a, v16h b, v8f c) {
  return __builtin_amdgcn_wmma_f32_16x16x32_f16(false, a, false, b,
                                                (short)0, c, false, false);
}

// Async copy 16B/lane directly into LDS (tracked with ASYNCcnt).
// Low 32 bits of a generic LDS pointer are the LDS byte offset (aperture map).
__device__ __forceinline__ void async_copy_b128(const f16* g, f16* l) {
  unsigned laddr = (unsigned)(unsigned long long)l;
  asm volatile("global_load_async_to_lds_b128 %0, %1, off"
               :: "v"(laddr), "v"(g) : "memory");
}
__device__ __forceinline__ void wait_async0() {
  asm volatile("s_wait_asynccnt 0" ::: "memory");
}
__device__ __forceinline__ void wait_ds0() {
  asm volatile("s_wait_dscnt 0" ::: "memory");
}

// A-matrix tile 16x32 (f16 source). Lane l: row = l&15,
// K = {0..7,16..23} (lanes 0-15) or {8..15,24..31} (lanes 16-31).
__device__ __forceinline__ v16h load_a_f16(const f16* base, int stride, int lane) {
  int row = lane & 15;
  int kb  = (lane < 16) ? 0 : 8;
  const f16* p = base + row * stride + kb;
  h8 lo = *(const h8*)p;
  h8 hi = *(const h8*)(p + 16);
  v16h a;
#pragma unroll
  for (int e = 0; e < 8; ++e) { a[e] = lo[e]; a[e + 8] = hi[e]; }
  return a;
}

// Same A layout, fp32 source with on-the-fly conversion.
__device__ __forceinline__ v16h load_a_f32(const float* base, int stride, int lane) {
  int row = lane & 15;
  int kb  = (lane < 16) ? 0 : 8;
  const float* p = base + row * stride + kb;
  v16h a;
#pragma unroll
  for (int e = 0; e < 8; ++e) {
    a[e]     = (_Float16)p[e];
    a[e + 8] = (_Float16)p[16 + e];
  }
  return a;
}

// B-matrix tile 32x16, K contiguous in memory. Lane l: col = l&15;
// lanes 0-15 hold K={0..7,16..23}, lanes 16-31 hold K={8..15,24..31}. Two 16B loads.
__device__ __forceinline__ v16h load_b_f16c(const f16* base, int cstride, int lane) {
  int col = lane & 15;
  int kb  = (lane < 16) ? 0 : 8;
  const f16* p = base + col * cstride + kb;
  h8 lo = *(const h8*)p;
  h8 hi = *(const h8*)(p + 16);
  v16h b;
#pragma unroll
  for (int e = 0; e < 8; ++e) { b[e] = lo[e]; b[e + 8] = hi[e]; }
  return b;
}

// ---------------------------------------------------------------------------
// 1) Weight transpose + fp32->f16 convert: Wt[out][in] = W[in][out]
// ---------------------------------------------------------------------------
__global__ void wt_convert_kernel(const float* __restrict__ W, f16* __restrict__ Wt) {
  int idx = blockIdx.x * 256 + threadIdx.x;        // out-major flat index
  int out = idx >> 9, in = idx & 511;
  Wt[idx] = (f16)W[(size_t)in * DMODEL + out];
}

// ---------------------------------------------------------------------------
// 2) Projection GEMM: Y[b][h][s][d] (f16) = X (fp32, rows x 512) @ Wt^T + bias
//    Block: 16 rows, 8 waves x 64 cols. K loop in steps of 32 (16 WMMA/acc).
// ---------------------------------------------------------------------------
__global__ void proj_kernel(const float* __restrict__ X, const f16* __restrict__ Wt,
                            const float* __restrict__ bias, f16* __restrict__ Y) {
  int tid = threadIdx.x, wave = tid >> 5, lane = tid & 31;
  int m0 = blockIdx.x * 16;
  int c0 = wave * 64;
  v8f acc[4] = {};
  for (int kk = 0; kk < DMODEL; kk += 32) {
    v16h a = load_a_f32(X + (size_t)m0 * DMODEL + kk, DMODEL, lane);
#pragma unroll
    for (int t = 0; t < 4; ++t) {
      v16h bt = load_b_f16c(Wt + (size_t)(c0 + t * 16) * DMODEL + kk, DMODEL, lane);
      acc[t] = wmma_f16(a, bt, acc[t]);
    }
  }
  int rbase = (lane < 16) ? 0 : 8;
#pragma unroll
  for (int t = 0; t < 4; ++t) {
    int c  = c0 + t * 16 + (lane & 15);
    float bv = bias[c];
    int hh = c >> 6, dd = c & 63;
#pragma unroll
    for (int r = 0; r < 8; ++r) {
      int m  = m0 + rbase + r;
      int bb = m >> 11, ss = m & (S1 - 1);
      Y[(((size_t)bb * HEADS + hh) * S1 + ss) * HDIM + dd] = (f16)(acc[t][r] + bv);
    }
  }
}

// ---------------------------------------------------------------------------
// 3) Flash cross-attention. Grid (S1/128, BS*HEADS), block 256 (8 waves).
//    32 keys / iteration: async-to-LDS double-buffered K/V tiles kept in two
//    layouts (row-major for QK^T, transposed for P.V). 8 WMMA / iteration /
//    wave. V == K (reference re-uses k as v). scale = 1/sqrt(512) (attn_dim!).
// ---------------------------------------------------------------------------
__global__ void attn_kernel(const f16* __restrict__ Qw, const f16* __restrict__ Kw,
                            const int* __restrict__ m1, const int* __restrict__ m2,
                            float* __restrict__ Ow) {
  __shared__ __attribute__((aligned(16))) f16 lds_k [2][32 * HDIM];  // [key][d]
  __shared__ __attribute__((aligned(16))) f16 lds_kt[2][HDIM * 32];  // [d][key]
  __shared__ __attribute__((aligned(16))) f16 lds_p [8][16 * 32];    // per-wave P

  int tid = threadIdx.x, wave = tid >> 5, lane = tid & 31;
  int bh = blockIdx.y;                      // b*HEADS + h
  int b  = bh >> 3, h = bh & 7;
  int qbase = blockIdx.x * 128 + wave * 16;

  const f16* Qp = Qw + ((size_t)bh * S1 + qbase) * HDIM;
  const f16* Kp = Kw + (size_t)bh * S2 * HDIM;

  // Per-thread staging chunk: 16B of K tile (32 keys x 64 d).
  int skey = tid >> 3, sdc = (tid & 7) * 8;

  // Q tile 16x64 -> two A tiles (d 0..31, 32..63), kept in registers.
  v16h qa0 = load_a_f16(Qp, HDIM, lane);
  v16h qa1 = load_a_f16(Qp + 32, HDIM, lane);

  int rbase = (lane < 16) ? 0 : 8;
  bool mrow[8];
#pragma unroll
  for (int r = 0; r < 8; ++r)
    mrow[r] = m1[b * S1 + qbase + rbase + r] != 0;

  float mmax[8], lsum[8];
  v8f o[4] = {};
#pragma unroll
  for (int r = 0; r < 8; ++r) { mmax[r] = -3.0e38f; lsum[r] = 0.f; }

  const float scale = 0.044194173824159216f;   // 1/sqrt(512)

  // Prime buffer 0: async copy, then build the transposed copy from the 16B
  // this thread itself deposited (no cross-thread dependency before barrier).
  async_copy_b128(&Kp[(size_t)skey * HDIM + sdc], &lds_k[0][skey * HDIM + sdc]);
  {
    wait_async0();
    h8 v = *(const h8*)&lds_k[0][skey * HDIM + sdc];
#pragma unroll
    for (int e = 0; e < 8; ++e) lds_kt[0][(sdc + e) * 32 + skey] = v[e];
  }
  __syncthreads();

  int buf = 0;
  for (int kt = 0; kt < S2; kt += 32) {
    int nb = buf ^ 1;
    bool more = (kt + 32) < S2;
    if (more)   // overlap next tile's DMA with this tile's math
      async_copy_b128(&Kp[(size_t)(kt + 32 + skey) * HDIM + sdc],
                      &lds_k[nb][skey * HDIM + sdc]);

    bool mk0 = m2[b * S2 + kt + (lane & 15)] != 0;
    bool mk1 = m2[b * S2 + kt + 16 + (lane & 15)] != 0;

    // S tiles: keys [kt,kt+16) and [kt+16,kt+32), reduction over d=64.
    v8f s0 = {}, s1 = {};
    s0 = wmma_f16(qa0, load_b_f16c(&lds_k[buf][0],              HDIM, lane), s0);
    s0 = wmma_f16(qa1, load_b_f16c(&lds_k[buf][32],             HDIM, lane), s0);
    s1 = wmma_f16(qa0, load_b_f16c(&lds_k[buf][16 * HDIM],      HDIM, lane), s1);
    s1 = wmma_f16(qa1, load_b_f16c(&lds_k[buf][16 * HDIM + 32], HDIM, lane), s1);

    // Masked scale + online softmax (rows live across 16 lanes in C layout).
#pragma unroll
    for (int r = 0; r < 8; ++r) {
      float t0 = s0[r] * scale + ((mrow[r] && mk0) ? 0.f : NEGV);
      float t1 = s1[r] * scale + ((mrow[r] && mk1) ? 0.f : NEGV);
      float rm = fmaxf(t0, t1);
#pragma unroll
      for (int m = 8; m >= 1; m >>= 1) rm = fmaxf(rm, __shfl_xor(rm, m, 32));
      float mn = fmaxf(mmax[r], rm);
      float alpha = __expf(mmax[r] - mn);
      mmax[r] = mn;
      float p0 = __expf(t0 - mn), p1 = __expf(t1 - mn);
      float rs = p0 + p1;
#pragma unroll
      for (int m = 8; m >= 1; m >>= 1) rs += __shfl_xor(rs, m, 32);
      lsum[r] = lsum[r] * alpha + rs;
#pragma unroll
      for (int t = 0; t < 4; ++t) o[t][r] *= alpha;
      // P back to LDS in row-major [q][key] for the A-layout reload.
      f16* pp = &lds_p[wave][(rbase + r) * 32 + (lane & 15)];
      pp[0]  = (f16)p0;
      pp[16] = (f16)p1;
    }
    wait_ds0();   // wave-internal LDS RAW on the P scratch

    // O += P (16x32) x V (32x64); V read contiguous from transposed tile.
    v16h pa = load_a_f16(&lds_p[wave][0], 32, lane);
#pragma unroll
    for (int t = 0; t < 4; ++t) {
      v16h bv = load_b_f16c(&lds_kt[buf][(t * 16) * 32], 32, lane);
      o[t] = wmma_f16(pa, bv, o[t]);
    }

    if (more) {   // land the async tile, mirror it into the transposed copy
      wait_async0();
      h8 v = *(const h8*)&lds_k[nb][skey * HDIM + sdc];
#pragma unroll
      for (int e = 0; e < 8; ++e) lds_kt[nb][(sdc + e) * 32 + skey] = v[e];
    }
    __syncthreads();   // single barrier per iteration (double-buffered)
    buf = nb;
  }

  // Normalize and store fp32 attn_res[b][s][h*64+d].
  float* Op = Ow + ((size_t)b * S1 + qbase) * DMODEL + h * HDIM;
#pragma unroll
  for (int r = 0; r < 8; ++r) {
    float inv = 1.f / lsum[r];
#pragma unroll
    for (int t = 0; t < 4; ++t)
      Op[(size_t)(rbase + r) * DMODEL + t * 16 + (lane & 15)] = o[t][r] * inv;
  }
}

// ---------------------------------------------------------------------------
// 4) Fused epilogue: out = LN2(attn + LN1(attn + modal1)). One row per block.
// ---------------------------------------------------------------------------
__device__ __forceinline__ float block_sum(float v, float* red, int tid) {
#pragma unroll
  for (int m = 16; m >= 1; m >>= 1) v += __shfl_xor(v, m, 32);
  __syncthreads();
  if ((tid & 31) == 0) red[tid >> 5] = v;
  __syncthreads();
  float s = 0.f;
#pragma unroll
  for (int i = 0; i < 8; ++i) s += red[i];
  return s;
}

__global__ void epilogue_kernel(const float* __restrict__ attn,
                                const float* __restrict__ modal1,
                                const float* __restrict__ g1, const float* __restrict__ b1,
                                const float* __restrict__ g2, const float* __restrict__ b2,
                                float* __restrict__ out) {
  __shared__ float red[8];
  int row = blockIdx.x, tid = threadIdx.x;
  const float* ap = attn  + (size_t)row * DMODEL;
  const float* mp = modal1 + (size_t)row * DMODEL;
  int i0 = tid, i1 = tid + 256;
  float a0 = ap[i0], a1 = ap[i1];
  float x0 = a0 + mp[i0], x1 = a1 + mp[i1];

  float mean = block_sum(x0 + x1, red, tid) * (1.f / 512.f);
  float d0 = x0 - mean, d1 = x1 - mean;
  float var = block_sum(d0 * d0 + d1 * d1, red, tid) * (1.f / 512.f);
  float rstd = rsqrtf(var + 1e-5f);
  float t0 = d0 * rstd * g1[i0] + b1[i0];
  float t1 = d1 * rstd * g1[i1] + b1[i1];

  float y0 = a0 + t0, y1 = a1 + t1;
  float mean2 = block_sum(y0 + y1, red, tid) * (1.f / 512.f);
  float e0 = y0 - mean2, e1 = y1 - mean2;
  float var2 = block_sum(e0 * e0 + e1 * e1, red, tid) * (1.f / 512.f);
  float rstd2 = rsqrtf(var2 + 1e-5f);

  out[(size_t)row * DMODEL + i0] = e0 * rstd2 * g2[i0] + b2[i0];
  out[(size_t)row * DMODEL + i1] = e1 * rstd2 * g2[i1] + b2[i1];
}

// ---------------------------------------------------------------------------
extern "C" void kernel_launch(void* const* d_in, const int* in_sizes, int n_in,
                              void* d_out, int out_size, void* d_ws, size_t ws_size,
                              hipStream_t stream) {
  (void)in_sizes; (void)n_in; (void)out_size; (void)ws_size;
  const float* modal1 = (const float*)d_in[0];
  const float* modal2 = (const float*)d_in[1];
  const int*   um1    = (const int*)  d_in[2];
  const int*   um2    = (const int*)  d_in[3];
  const float* Wq     = (const float*)d_in[4];
  const float* bq     = (const float*)d_in[5];
  const float* Wkv    = (const float*)d_in[6];
  const float* bkv    = (const float*)d_in[7];
  const float* ln1g   = (const float*)d_in[8];
  const float* ln1b   = (const float*)d_in[9];
  // d_in[10..13] = W1,b1,W2,b2 : dead FFN, intentionally unused.
  const float* ln2g   = (const float*)d_in[14];
  const float* ln2b   = (const float*)d_in[15];
  float* out = (float*)d_out;

  // Workspace carve (~33 MB total).
  f16* wtq  = (f16*)d_ws;                                  // 512*512 f16
  f16* wtkv = wtq + (size_t)DMODEL * DMODEL;
  f16* qws  = wtkv + (size_t)DMODEL * DMODEL;              // [b][h][s][d]
  f16* kws  = qws + (size_t)BS * HEADS * S1 * HDIM;
  float* aws = (float*)(kws + (size_t)BS * HEADS * S2 * HDIM);  // fp32 attn_res

  wt_convert_kernel<<<(DMODEL * DMODEL) / 256, 256, 0, stream>>>(Wq, wtq);
  wt_convert_kernel<<<(DMODEL * DMODEL) / 256, 256, 0, stream>>>(Wkv, wtkv);

  proj_kernel<<<(BS * S1) / 16, 256, 0, stream>>>(modal1, wtq, bq, qws);
  proj_kernel<<<(BS * S2) / 16, 256, 0, stream>>>(modal2, wtkv, bkv, kws);

  attn_kernel<<<dim3(S1 / 128, BS * HEADS), 256, 0, stream>>>(qws, kws, um1, um2, aws);

  epilogue_kernel<<<BS * S1, 256, 0, stream>>>(aws, modal1, ln1g, ln1b, ln2g, ln2b, out);
}
